// HierarchicalMambaBlock_79577154060423
// MI455X (gfx1250) — matrix operand order repaired and checked by reading, not compiled, not run on hardware
//
#include <hip/hip_runtime.h>
#include <hip/hip_bf16.h>
#include <stdint.h>

// ---------------------------------------------------------------------------
// Hierarchical Mamba block for MI455X (gfx1250, wave32).
// GEMMs: bf16 v_wmma_f32_16x16x32_bf16, f32 accum, double-buffered LDS tiles
// filled by gfx1250 async-DMA (global_load_async_to_lds_b128 / ASYNCcnt).
// Weights are pre-transposed to [N][K] so both operands stage contiguously.
// ---------------------------------------------------------------------------

typedef __attribute__((ext_vector_type(16))) __bf16 v16bf;
typedef __attribute__((ext_vector_type(8)))  float  v8f;

#define TSEQ 2048
#define DIMM 768
#define DI   1536
#define NSC  3

static __device__ __forceinline__ float sigmoidf_(float x) { return 1.0f / (1.0f + __expf(-x)); }
static __device__ __forceinline__ float siluf_(float x)    { return x * sigmoidf_(x); }
static __device__ __forceinline__ float softplusf_(float x){ return (x > 20.0f) ? x : log1pf(__expf(x)); }

// ---------------------------------------------------------------------------
// f32 -> bf16 conversion
// ---------------------------------------------------------------------------
__global__ void f2bf_kernel(const float* __restrict__ in, __bf16* __restrict__ out, long n) {
  long i = (long)blockIdx.x * blockDim.x + threadIdx.x;
  long stride = (long)gridDim.x * blockDim.x;
  for (; i < n; i += stride) out[i] = (__bf16)in[i];
}

// f32 [K][N] -> bf16 transposed [N][K]
__global__ void f2bf_t_kernel(const float* __restrict__ in, __bf16* __restrict__ out,
                              int K, int N) {
  long total = (long)K * N;
  long i = (long)blockIdx.x * blockDim.x + threadIdx.x;
  long gs = (long)gridDim.x * blockDim.x;
  for (; i < total; i += gs) {
    int n = (int)(i / K), k = (int)(i % K);
    out[i] = (__bf16)in[(size_t)k * N + n];
  }
}

// ---------------------------------------------------------------------------
// bf16 WMMA GEMM: C[MxN] = act(A[MxK] @ Bt[NxK]^T) (+ addend)
// Block tile 128x128, K-step 32, 8 waves (4M x 2N), 32x64 per wave.
// ---------------------------------------------------------------------------
#define BM 128
#define BN 128
#define BK 32
#define APITCH 40   // bf16 elems per LDS row (32 + 8 pad); 80 B rows, 16B-aligned
#define BPITCH 40

union Frag16 { v16bf v; uint4 q[2]; };

__global__ __launch_bounds__(256)
void gemm_bf16_wmma(const __bf16* __restrict__ A, int lda,
                    const __bf16* __restrict__ Bt, int ldbt,
                    float* __restrict__ C, int ldc,
                    int M, int N, int K,
                    int act, const float* __restrict__ addend, int ldadd) {
  __shared__ __align__(16) __bf16 As[2][BM * APITCH];
  __shared__ __align__(16) __bf16 Bs[2][BN * BPITCH];

  const int tid  = threadIdx.x;
  const int lane = tid & 31;
  const int wave = tid >> 5;
  const int wm   = wave >> 1;   // 0..3 : 32-row strip
  const int wn   = wave & 1;    // 0..1 : 64-col strip
  const int l16  = lane & 15;
  const int lh   = lane >> 4;

  const int bm0 = blockIdx.y * BM;
  const int bn0 = blockIdx.x * BN;

  v8f acc[2][4] = {};

  // --- async DMA staging: 4 x global_load_async_to_lds_b128 per thread ---
  auto stage_async = [&](int k0, int buf) {
    #pragma unroll
    for (int i = 0; i < 2; ++i) {
      int c    = tid + i * 256;      // 0..511
      int row  = c >> 2;             // 0..127
      int col8 = (c & 3) * 8;        // 0,8,16,24
      unsigned la = (unsigned)(uintptr_t)(&As[buf][row * APITCH + col8]);
      unsigned long long ga =
          (unsigned long long)(uintptr_t)(A + (size_t)(bm0 + row) * lda + k0 + col8);
      asm volatile("global_load_async_to_lds_b128 %0, %1, off"
                   :: "v"(la), "v"(ga) : "memory");
      unsigned lb = (unsigned)(uintptr_t)(&Bs[buf][row * BPITCH + col8]);
      unsigned long long gb =
          (unsigned long long)(uintptr_t)(Bt + (size_t)(bn0 + row) * ldbt + k0 + col8);
      asm volatile("global_load_async_to_lds_b128 %0, %1, off"
                   :: "v"(lb), "v"(gb) : "memory");
    }
  };

  // --- synchronous guarded staging (ragged tiles), zero-filled ---
  auto stage_sync = [&](int k0, int buf) {
    #pragma unroll
    for (int i = 0; i < 2; ++i) {
      int c    = tid + i * 256;
      int row  = c >> 2;
      int col8 = (c & 3) * 8;
      uint4 va = make_uint4(0u, 0u, 0u, 0u);
      if (bm0 + row < M)
        va = *(const uint4*)(A + (size_t)(bm0 + row) * lda + k0 + col8);
      *(uint4*)(&As[buf][row * APITCH + col8]) = va;
      uint4 vb = make_uint4(0u, 0u, 0u, 0u);
      if (bn0 + row < N)
        vb = *(const uint4*)(Bt + (size_t)(bn0 + row) * ldbt + k0 + col8);
      *(uint4*)(&Bs[buf][row * BPITCH + col8]) = vb;
    }
  };

  // --- one K-step of WMMAs from LDS buffer `buf` ---
  auto compute_tile = [&](int buf) {
    Frag16 afr[2], bfr[4];
    #pragma unroll
    for (int tm = 0; tm < 2; ++tm) {
      int row = wm * 32 + tm * 16 + l16;
      const __bf16* p = &As[buf][row * APITCH + lh * 8];
      afr[tm].q[0] = *(const uint4*)p;        // K = lh*8 .. +7
      afr[tm].q[1] = *(const uint4*)(p + 16); // K = 16+lh*8 .. +7
    }
    #pragma unroll
    for (int tn = 0; tn < 4; ++tn) {
      int row = wn * 64 + tn * 16 + l16;      // B column index n
      const __bf16* p = &Bs[buf][row * BPITCH + lh * 8];
      bfr[tn].q[0] = *(const uint4*)p;        // K = lh*8 .. +7
      bfr[tn].q[1] = *(const uint4*)(p + 16); // K = 16+lh*8 .. +7
    }
    #pragma unroll
    for (int tm = 0; tm < 2; ++tm)
      #pragma unroll
      for (int tn = 0; tn < 4; ++tn)
        acc[tm][tn] = __builtin_amdgcn_wmma_f32_16x16x32_bf16(
            false, afr[tm].v, false, bfr[tn].v,
            (short)0, acc[tm][tn], false, false);
  };

  const int nk = K / BK;
  const bool full = (bm0 + BM <= M) && (bn0 + BN <= N);

  if (full) {
    // double-buffered async pipeline
    stage_async(0, 0);
    for (int kt = 0; kt < nk; ++kt) {
      int cb = kt & 1;
      if (kt + 1 < nk) {
        stage_async((kt + 1) * BK, cb ^ 1);
        // 4 newer loads outstanding; in-order completion => tile kt landed
        asm volatile("s_wait_asynccnt %0" :: "i"(4));
      } else {
        asm volatile("s_wait_asynccnt %0" :: "i"(0));
      }
      __syncthreads();
      compute_tile(cb);
      __syncthreads();
    }
  } else {
    for (int kt = 0; kt < nk; ++kt) {
      stage_sync(kt * BK, 0);
      __syncthreads();
      compute_tile(0);
      __syncthreads();
    }
  }

  // ---- epilogue: D layout -> VGPR r holds M = r + 8*lh, N = l16 ----
  #pragma unroll
  for (int tm = 0; tm < 2; ++tm) {
    #pragma unroll
    for (int tn = 0; tn < 4; ++tn) {
      int col = bn0 + wn * 64 + tn * 16 + l16;
      if (col >= N) continue;
      #pragma unroll
      for (int r = 0; r < 8; ++r) {
        int row = bm0 + wm * 32 + tm * 16 + r + lh * 8;
        if (row >= M) continue;
        float v = acc[tm][tn][r];
        if (act == 1) v = siluf_(v);
        if (addend) v += addend[(size_t)row * ldadd + col];
        C[(size_t)row * ldc + col] = v;
      }
    }
  }
}

// ---------------------------------------------------------------------------
// Downsample (mean over `stride` steps); de-strides x_in view out of xz.
// ---------------------------------------------------------------------------
__global__ void downsample_kernel(const float* __restrict__ in, int ld_in,
                                  float* __restrict__ out, int T_out, int stride, int D) {
  long total = (long)T_out * D;
  long i = (long)blockIdx.x * blockDim.x + threadIdx.x;
  long gs = (long)gridDim.x * blockDim.x;
  float inv = 1.0f / (float)stride;
  for (; i < total; i += gs) {
    int t = (int)(i / D), d = (int)(i % D);
    float s = 0.0f;
    for (int j = 0; j < stride; ++j)
      s += in[(size_t)(t * stride + j) * ld_in + d];
    out[i] = s * inv;
  }
}

// ---------------------------------------------------------------------------
// Causal depthwise conv (K=4) + bias + SiLU.  w layout [D][4].
// ---------------------------------------------------------------------------
__global__ void conv_silu_kernel(const float* __restrict__ x, const float* __restrict__ w,
                                 const float* __restrict__ b, float* __restrict__ y,
                                 int T, int D) {
  long total = (long)T * D;
  long i = (long)blockIdx.x * blockDim.x + threadIdx.x;
  long gs = (long)gridDim.x * blockDim.x;
  for (; i < total; i += gs) {
    int t = (int)(i / D), d = (int)(i % D);
    float acc = b[d];
    #pragma unroll
    for (int k = 0; k < 4; ++k) {
      int ts = t + k - 3;
      if (ts >= 0) acc += x[(size_t)ts * D + d] * w[d * 4 + k];
    }
    y[i] = siluf_(acc);
  }
}

// ---------------------------------------------------------------------------
// dt = softplus(softplus(B_ssm @ dtw + dtb)).  proj row: [B(16) | C(16)].
// ---------------------------------------------------------------------------
__global__ void dt_kernel(const float* __restrict__ proj, const float* __restrict__ dtw,
                          const float* __restrict__ dtb, float* __restrict__ dt,
                          int T, int D) {
  long total = (long)T * D;
  long i = (long)blockIdx.x * blockDim.x + threadIdx.x;
  long gs = (long)gridDim.x * blockDim.x;
  for (; i < total; i += gs) {
    int t = (int)(i / D), d = (int)(i % D);
    float a = dtb[d];
    #pragma unroll
    for (int n = 0; n < 16; ++n)
      a += proj[t * 32 + n] * dtw[n * D + d];
    dt[i] = softplusf_(softplusf_(a));
  }
}

// ---------------------------------------------------------------------------
// Sequential SSM scan. Block = 256 threads = 16 channels (d) x 16 states (n).
// h_t = exp(dt*A_n)*h_{t-1} + dt*B_n*xc ;  y = sum_n C_n*h_t + D_d*xc
// ---------------------------------------------------------------------------
__global__ __launch_bounds__(256)
void ssm_scan_kernel(const float* __restrict__ dt, const float* __restrict__ proj,
                     const float* __restrict__ xc, const float* __restrict__ Dp,
                     float* __restrict__ y, int T, int D) {
  int n = threadIdx.x & 15;
  int d = blockIdx.x * 16 + (threadIdx.x >> 4);
  float An = -(float)(n + 1);
  float Dd = Dp[d];
  float h = 0.0f;
  for (int t = 0; t < T; ++t) {
    float dtv = dt[(size_t)t * D + d];
    float xcv = xc[(size_t)t * D + d];
    float bv  = proj[t * 32 + n];
    float cv  = proj[t * 32 + 16 + n];
    h = __expf(dtv * An) * h + dtv * bv * xcv;
    float c = cv * h;
    c += __shfl_xor(c, 8, 16);
    c += __shfl_xor(c, 4, 16);
    c += __shfl_xor(c, 2, 16);
    c += __shfl_xor(c, 1, 16);
    if (n == 0) y[(size_t)t * D + d] = c + Dd * xcv;
  }
}

// ---------------------------------------------------------------------------
// Linear upsample + softmax-weighted fuse + mean ctx.
// ---------------------------------------------------------------------------
static __device__ __forceinline__ float upsample1(const float* __restrict__ ys, int Ts,
                                                  int Tt, int t, int d, int D) {
  float scale = (float)Ts / (float)Tt;
  float pos = ((float)t + 0.5f) * scale - 0.5f;
  pos = fminf(fmaxf(pos, 0.0f), (float)(Ts - 1));
  int lo = (int)floorf(pos);
  int hi = min(lo + 1, Ts - 1);
  float w = pos - (float)lo;
  return ys[(size_t)lo * D + d] * (1.0f - w) + ys[(size_t)hi * D + d] * w;
}

__global__ void combine_kernel(const float* __restrict__ y0, const float* __restrict__ y1,
                               const float* __restrict__ y2, const float* __restrict__ sw,
                               float* __restrict__ fused, float* __restrict__ ctx,
                               int T, int D) {
  float w0 = sw[0], w1 = sw[1], w2 = sw[2];
  float m = fmaxf(w0, fmaxf(w1, w2));
  float e0 = __expf(w0 - m), e1 = __expf(w1 - m), e2 = __expf(w2 - m);
  float inv = 1.0f / (e0 + e1 + e2);
  e0 *= inv; e1 *= inv; e2 *= inv;
  long total = (long)T * D;
  long i = (long)blockIdx.x * blockDim.x + threadIdx.x;
  long gs = (long)gridDim.x * blockDim.x;
  for (; i < total; i += gs) {
    int t = (int)(i / D), d = (int)(i % D);
    float v0 = y0[i];
    float v1 = upsample1(y1, T / 2, T, t, d, D);
    float v2 = upsample1(y2, T / 4, T, t, d, D);
    fused[i] = e0 * v0 + e1 * v1 + e2 * v2;
    ctx[i]   = (v0 + v1 + v2) * (1.0f / 3.0f);
  }
}

// ---------------------------------------------------------------------------
// out = fused * sigmoid(gpre) * silu(gate)   (gate = xz[:, 1536:])
// ---------------------------------------------------------------------------
__global__ void gate_mul_kernel(const float* __restrict__ fused, const float* __restrict__ gpre,
                                const float* __restrict__ xz, float* __restrict__ out,
                                int T, int D) {
  long total = (long)T * D;
  long i = (long)blockIdx.x * blockDim.x + threadIdx.x;
  long gs = (long)gridDim.x * blockDim.x;
  for (; i < total; i += gs) {
    int t = (int)(i / D), d = (int)(i % D);
    float gt = xz[(size_t)t * (2 * DI) + DI + d];
    out[i] = fused[i] * sigmoidf_(gpre[i]) * siluf_(gt);
  }
}

// ---------------------------------------------------------------------------
// LayerNorm over D=768, one block per row.
// ---------------------------------------------------------------------------
__global__ __launch_bounds__(256)
void layernorm_kernel(const float* __restrict__ ypre, const float* __restrict__ gamma,
                      const float* __restrict__ beta, float* __restrict__ out, int D) {
  int t = blockIdx.x;
  const float* row = ypre + (size_t)t * D;
  __shared__ float red[8];

  float s = 0.0f;
  for (int d = threadIdx.x; d < D; d += 256) s += row[d];
  for (int off = 16; off; off >>= 1) s += __shfl_xor(s, off, 32);
  if ((threadIdx.x & 31) == 0) red[threadIdx.x >> 5] = s;
  __syncthreads();
  float mean = 0.0f;
  #pragma unroll
  for (int i = 0; i < 8; ++i) mean += red[i];
  mean /= (float)D;
  __syncthreads();

  float v = 0.0f;
  for (int d = threadIdx.x; d < D; d += 256) {
    float z = row[d] - mean;
    v += z * z;
  }
  for (int off = 16; off; off >>= 1) v += __shfl_xor(v, off, 32);
  if ((threadIdx.x & 31) == 0) red[threadIdx.x >> 5] = v;
  __syncthreads();
  float var = 0.0f;
  #pragma unroll
  for (int i = 0; i < 8; ++i) var += red[i];
  var /= (float)D;
  float inv = rsqrtf(var + 1e-5f);

  for (int d = threadIdx.x; d < D; d += 256)
    out[(size_t)t * D + d] = (row[d] - mean) * inv * gamma[d] + beta[d];
}

// ---------------------------------------------------------------------------
// Host orchestration
// ---------------------------------------------------------------------------
extern "C" void kernel_launch(void* const* d_in, const int* in_sizes, int n_in,
                              void* d_out, int out_size, void* d_ws, size_t ws_size,
                              hipStream_t stream) {
  (void)in_sizes; (void)n_in; (void)out_size; (void)ws_size;

  const float* x        = (const float*)d_in[0];
  const float* in_proj  = (const float*)d_in[1];
  const float* conv_w   = (const float*)d_in[2];
  const float* conv_b   = (const float*)d_in[3];
  const float* xproj_w  = (const float*)d_in[4];
  const float* dtproj_w = (const float*)d_in[5];
  const float* dtproj_b = (const float*)d_in[6];
  const float* D_param  = (const float*)d_in[7];
  const float* scale_w  = (const float*)d_in[8];
  const float* gate1_w  = (const float*)d_in[9];
  const float* gate2_w  = (const float*)d_in[10];
  const float* out_proj = (const float*)d_in[11];
  const float* ln_gamma = (const float*)d_in[12];
  const float* ln_beta  = (const float*)d_in[13];
  float* out = (float*)d_out;

  char* ws = (char*)d_ws;
  size_t cur = 0;
  auto alloc = [&](size_t bytes) -> void* {
    void* p = ws + cur;
    cur += (bytes + 255) & ~(size_t)255;
    return p;
  };

  __bf16* xb         = (__bf16*)alloc(sizeof(__bf16) * (size_t)TSEQ * DIMM);
  __bf16* wb_inprojT = (__bf16*)alloc(sizeof(__bf16) * (size_t)(2 * DI) * DIMM);
  __bf16* wb_xprojT  = (__bf16*)alloc(sizeof(__bf16) * (size_t)NSC * 32 * DI);
  __bf16* wb_gate1T  = (__bf16*)alloc(sizeof(__bf16) * (size_t)(DI / 2) * DI);
  __bf16* wb_gate2T  = (__bf16*)alloc(sizeof(__bf16) * (size_t)DI * (DI / 2));
  __bf16* wb_outpT   = (__bf16*)alloc(sizeof(__bf16) * (size_t)DIMM * DI);

  float*  xz    = (float*)alloc(sizeof(float) * (size_t)TSEQ * 2 * DI);
  float*  ybuf0 = (float*)alloc(sizeof(float) * (size_t)TSEQ * DI);
  float*  ybuf1 = (float*)alloc(sizeof(float) * (size_t)(TSEQ / 2) * DI);
  float*  ybuf2 = (float*)alloc(sizeof(float) * (size_t)(TSEQ / 4) * DI);
  float*  fused = (float*)alloc(sizeof(float) * (size_t)TSEQ * DI);
  float*  ctx   = (float*)alloc(sizeof(float) * (size_t)TSEQ * DI);
  float*  xs    = (float*)alloc(sizeof(float) * (size_t)TSEQ * DI);
  float*  xc    = (float*)alloc(sizeof(float) * (size_t)TSEQ * DI);
  __bf16* xcb   = (__bf16*)alloc(sizeof(__bf16) * (size_t)TSEQ * DI);
  float*  proj  = (float*)alloc(sizeof(float) * (size_t)TSEQ * 32);
  float*  dtb_  = (float*)alloc(sizeof(float) * (size_t)TSEQ * DI);
  __bf16* ctxb  = (__bf16*)alloc(sizeof(__bf16) * (size_t)TSEQ * DI);
  float*  tmp1  = (float*)alloc(sizeof(float) * (size_t)TSEQ * (DI / 2));
  __bf16* tmp1b = (__bf16*)alloc(sizeof(__bf16) * (size_t)TSEQ * (DI / 2));
  float*  gpre  = (float*)alloc(sizeof(float) * (size_t)TSEQ * DI);
  float*  outpre  = (float*)alloc(sizeof(float) * (size_t)TSEQ * DI);
  __bf16* outpreb = (__bf16*)alloc(sizeof(__bf16) * (size_t)TSEQ * DI);
  float*  ypre  = (float*)alloc(sizeof(float) * (size_t)TSEQ * DIMM);

  auto grid1 = [](long n) { return dim3((unsigned)((n + 255) / 256)); };
  auto gemm = [&](const __bf16* A, int lda, const __bf16* Bt, int ldbt,
                  float* C, int ldc, int M, int N, int K, int act,
                  const float* add, int ldadd) {
    dim3 g((unsigned)((N + BN - 1) / BN), (unsigned)((M + BM - 1) / BM));
    gemm_bf16_wmma<<<g, 256, 0, stream>>>(A, lda, Bt, ldbt, C, ldc, M, N, K, act, add, ldadd);
  };

  // --- conversions: activations plain, weights transposed to [N][K] bf16 ---
  f2bf_kernel<<<grid1((long)TSEQ * DIMM), 256, 0, stream>>>(x, xb, (long)TSEQ * DIMM);
  f2bf_t_kernel<<<grid1((long)DIMM * 2 * DI), 256, 0, stream>>>(in_proj, wb_inprojT, DIMM, 2 * DI);
  for (int s = 0; s < NSC; ++s)
    f2bf_t_kernel<<<grid1((long)DI * 32), 256, 0, stream>>>(
        xproj_w + (size_t)s * DI * 32, wb_xprojT + (size_t)s * 32 * DI, DI, 32);
  f2bf_t_kernel<<<grid1((long)DI * (DI / 2)), 256, 0, stream>>>(gate1_w, wb_gate1T, DI, DI / 2);
  f2bf_t_kernel<<<grid1((long)(DI / 2) * DI), 256, 0, stream>>>(gate2_w, wb_gate2T, DI / 2, DI);
  f2bf_t_kernel<<<grid1((long)DI * DIMM), 256, 0, stream>>>(out_proj, wb_outpT, DI, DIMM);

  // --- input projection: xz[T, 3072] = x @ in_proj_w ---
  gemm(xb, DIMM, wb_inprojT, DIMM, xz, 2 * DI, TSEQ, 2 * DI, DIMM, 0, nullptr, 0);

  // --- per-scale SSM ---
  float* ybufs[NSC] = { ybuf0, ybuf1, ybuf2 };
  for (int s = 0; s < NSC; ++s) {
    int strd = 1 << s;
    int Ts = TSEQ >> s;
    downsample_kernel<<<grid1((long)Ts * DI), 256, 0, stream>>>(xz, 2 * DI, xs, Ts, strd, DI);
    conv_silu_kernel<<<grid1((long)Ts * DI), 256, 0, stream>>>(
        xs, conv_w + (size_t)s * DI * 4, conv_b + (size_t)s * DI, xc, Ts, DI);
    f2bf_kernel<<<grid1((long)Ts * DI), 256, 0, stream>>>(xc, xcb, (long)Ts * DI);
    gemm(xcb, DI, wb_xprojT + (size_t)s * 32 * DI, DI, proj, 32, Ts, 32, DI, 0, nullptr, 0);
    dt_kernel<<<grid1((long)Ts * DI), 256, 0, stream>>>(
        proj, dtproj_w + (size_t)s * 16 * DI, dtproj_b + (size_t)s * DI, dtb_, Ts, DI);
    ssm_scan_kernel<<<dim3(DI / 16), 256, 0, stream>>>(
        dtb_, proj, xc, D_param + (size_t)s * DI, ybufs[s], Ts, DI);
  }

  // --- upsample + fuse ---
  combine_kernel<<<grid1((long)TSEQ * DI), 256, 0, stream>>>(
      ybuf0, ybuf1, ybuf2, scale_w, fused, ctx, TSEQ, DI);

  // --- gating MLP ---
  f2bf_kernel<<<grid1((long)TSEQ * DI), 256, 0, stream>>>(ctx, ctxb, (long)TSEQ * DI);
  gemm(ctxb, DI, wb_gate1T, DI, tmp1, DI / 2, TSEQ, DI / 2, DI, 1 /*silu*/, nullptr, 0);
  f2bf_kernel<<<grid1((long)TSEQ * (DI / 2)), 256, 0, stream>>>(tmp1, tmp1b, (long)TSEQ * (DI / 2));
  gemm(tmp1b, DI / 2, wb_gate2T, DI / 2, gpre, DI, TSEQ, DI, DI / 2, 0, nullptr, 0);
  gate_mul_kernel<<<grid1((long)TSEQ * DI), 256, 0, stream>>>(fused, gpre, xz, outpre, TSEQ, DI);

  // --- output projection + residual ---
  f2bf_kernel<<<grid1((long)TSEQ * DI), 256, 0, stream>>>(outpre, outpreb, (long)TSEQ * DI);
  gemm(outpreb, DI, wb_outpT, DI, ypre, DIMM, TSEQ, DIMM, DI, 0, x, DIMM);

  // --- LayerNorm ---
  layernorm_kernel<<<dim3(TSEQ), 256, 0, stream>>>(ypre, ln_gamma, ln_beta, out, DIMM);
}
